// RealSpaceEGNN_19215683682943
// MI455X (gfx1250) — matrix operand order repaired and checked by reading, not compiled, not run on hardware
//
#include <hip/hip_runtime.h>
#include <hip/hip_bf16.h>

typedef __attribute__((ext_vector_type(16))) _Float16 v16h;
typedef __attribute__((ext_vector_type(8)))  float    v8f;

static constexpr int NNODES = 10000;
static constexpr int NEDGES = 40000;
static constexpr int NGRAPH = 64;
static constexpr int HDIM   = 120;   // 32*1 + 16*3 + 8*5

// ---------------------------------------------------------------------------
// Wigner-3j (real basis) init kernel: replicates the Python CG + c2r transform
// ---------------------------------------------------------------------------
__device__ inline double dfact(int n) {
    const double f[9] = {1,1,2,6,24,120,720,5040,40320};
    return f[n];
}
__device__ inline double dtri(int a,int b,int c) {
    return dfact(a+b-c)*dfact(a-b+c)*dfact(-a+b+c)/dfact(a+b+c+1);
}
__device__ inline int imin3(int a,int b,int c){ int m=a<b?a:b; return m<c?m:c; }

__device__ double dcg(int j1,int m1,int j2,int m2,int j3,int m3) {
    if (m1+m2 != m3) return 0.0;
    int dj = j1>j2 ? j1-j2 : j2-j1;
    if (j3 < dj || j3 > j1+j2) return 0.0;
    double pre = sqrt((2.0*j3+1.0)*dtri(j1,j2,j3)*dfact(j3+m3)*dfact(j3-m3)*
                      dfact(j1-m1)*dfact(j1+m1)*dfact(j2-m2)*dfact(j2+m2));
    double s = 0.0;
    int kmax = imin3(j1+j2-j3, j1-m1, j2+m2);
    for (int k=0; k<=kmax; k++) {
        if (j3-j2+m1+k < 0 || j3-j1-m2+k < 0) continue;
        double den = dfact(k)*dfact(j1+j2-j3-k)*dfact(j1-m1-k)*dfact(j2+m2-k)*
                     dfact(j3-j2+m1+k)*dfact(j3-j1-m2+k);
        s += ((k&1) ? -1.0 : 1.0)/den;
    }
    return pre*s;
}

__device__ void c2r_fill(int l, double Ur[5][5], double Ui[5][5]) {
    for (int a=0;a<5;a++) for (int b=0;b<5;b++){ Ur[a][b]=0.0; Ui[a][b]=0.0; }
    const double is2 = 0.70710678118654752440;
    for (int m=-l; m<=l; m++) {
        int r = l+m;
        if (m == 0) { Ur[l][l] = 1.0; }
        else if (m > 0) {
            Ur[r][l+m] = ((m&1)?-1.0:1.0)*is2;
            Ur[r][l-m] = is2;
        } else {
            int p = -m;
            Ui[r][l+m] = is2;                       // i/sqrt2
            Ui[r][l-m] = -(((p&1)?-1.0:1.0))*is2;   // -i*(-1)^p/sqrt2
        }
    }
}

__global__ void w3j_init_kernel(float* w3j) {
    int t = threadIdx.x;
    if (t >= 11) return;
    const int l1s[11]={0,0,0,1,1,1,1,2,2,2,2};
    const int l2s[11]={0,1,2,0,1,1,2,0,1,2,2};
    const int l3s[11]={0,1,2,1,0,2,1,2,1,0,2};
    const int offs[11]={0,1,10,35,44,53,98,143,168,213,238};
    int l1=l1s[t], l2=l2s[t], l3=l3s[t];
    int d1=2*l1+1, d2=2*l2+1, d3=2*l3+1;

    double U1r[5][5],U1i[5][5],U2r[5][5],U2i[5][5],U3r[5][5],U3i[5][5];
    c2r_fill(l1,U1r,U1i); c2r_fill(l2,U2r,U2i); c2r_fill(l3,U3r,U3i);

    double Cc[5][5][5];
    for (int a=0;a<5;a++)for(int b=0;b<5;b++)for(int c=0;c<5;c++) Cc[a][b][c]=0.0;
    for (int m1=-l1;m1<=l1;m1++)
        for (int m2=-l2;m2<=l2;m2++) {
            int m3=m1+m2;
            if (m3>=-l3 && m3<=l3)
                Cc[l1+m1][l2+m2][l3+m3] = dcg(l1,m1,l2,m2,l3,m3);
        }
    double tr[125], ti[125];
    double sr=0.0, si=0.0;
    for (int i=0;i<d1;i++)
    for (int j=0;j<d2;j++)
    for (int k=0;k<d3;k++) {
        double ar=0.0, ai=0.0;
        for (int m=0;m<d1;m++) {
            double u1r=U1r[i][m], u1i=U1i[i][m];
            if (u1r==0.0 && u1i==0.0) continue;
            for (int nn=0;nn<d2;nn++) {
                double u2r=U2r[j][nn], u2i=U2i[j][nn];
                if (u2r==0.0 && u2i==0.0) continue;
                double t12r = u1r*u2r - u1i*u2i;
                double t12i = u1r*u2i + u1i*u2r;
                for (int o=0;o<d3;o++) {
                    double cc = Cc[m][nn][o];
                    if (cc==0.0) continue;
                    double u3r=U3r[k][o], u3i=U3i[k][o];
                    ar += (t12r*u3r + t12i*u3i)*cc;  // * conj(u3)
                    ai += (t12i*u3r - t12r*u3i)*cc;
                }
            }
        }
        int e = (i*d2+j)*d3+k;
        tr[e]=ar; ti[e]=ai;
        sr += fabs(ar); si += fabs(ai);
    }
    int sz=d1*d2*d3;
    for (int e=0;e<sz;e++)
        w3j[offs[t]+e] = (float)((sr>=si) ? tr[e] : ti[e]);
}

// ---------------------------------------------------------------------------
// Simple utility kernels
// ---------------------------------------------------------------------------
__global__ void zero_kernel(float* p, int n) {
    int i = blockIdx.x*blockDim.x + threadIdx.x;
    if (i < n) p[i] = 0.0f;
}

__global__ void edge_attr_kernel(const float* pos, const int* ei, float* ea) {
    int e = blockIdx.x*blockDim.x + threadIdx.x;
    if (e >= NEDGES) return;
    int r = ei[e], c = ei[NEDGES + e];
    float dx = pos[r*3+0]-pos[c*3+0];
    float dy = pos[r*3+1]-pos[c*3+1];
    float dz = pos[r*3+2]-pos[c*3+2];
    float d = sqrtf(dx*dx+dy*dy+dz*dz);
    float inv = 1.0f/(d+1e-8f);
    ea[e*4+0]=dx*inv; ea[e*4+1]=dy*inv; ea[e*4+2]=dz*inv; ea[e*4+3]=d;
}

// ---------------------------------------------------------------------------
// WMMA fragment helpers (exact tiling; no bounds checks).
// A frag: a[h] = rowptr[k],  k = h + 16*hi  -> two runs of 8 floats
// B frag: b[h] = B[(kk + h + 16*hi)*ldb + col]
// D frag: d[r] -> row r + 8*hi, col lane&15
// ---------------------------------------------------------------------------
__device__ inline v16h load_a_cols32(const float* __restrict__ rowptr, int hi) {
    const float4* p0 = reinterpret_cast<const float4*>(rowptr + (hi<<4));
    const float4* p1 = reinterpret_cast<const float4*>(rowptr + 8 + (hi<<4));
    float4 x0 = p0[0], x1 = p0[1], x2 = p1[0], x3 = p1[1];
    v16h a;
    a[0]=(_Float16)x0.x;  a[1]=(_Float16)x0.y;  a[2]=(_Float16)x0.z;  a[3]=(_Float16)x0.w;
    a[4]=(_Float16)x1.x;  a[5]=(_Float16)x1.y;  a[6]=(_Float16)x1.z;  a[7]=(_Float16)x1.w;
    a[8]=(_Float16)x2.x;  a[9]=(_Float16)x2.y;  a[10]=(_Float16)x2.z; a[11]=(_Float16)x2.w;
    a[12]=(_Float16)x3.x; a[13]=(_Float16)x3.y; a[14]=(_Float16)x3.z; a[15]=(_Float16)x3.w;
    return a;
}

__device__ inline v16h load_b_k32(const float* __restrict__ B, int ldb,
                                  int kk, int col, int hi) {
    float t[16];
#pragma unroll
    for (int h = 0; h < 16; h++) t[h] = B[(kk + h + (hi<<4))*ldb + col];
    v16h b;
#pragma unroll
    for (int h = 0; h < 16; h++) b[h] = (_Float16)t[h];
    return b;
}

// ---------------------------------------------------------------------------
// WMMA GEMM: C = scale * A(f32->f16) @ B(f32->f16); requires M%16==0,
// K%32==0, N%16==0. One wave per 16x16 tile; grid.x = (M/16)*(N/16).
// ---------------------------------------------------------------------------
__global__ void __launch_bounds__(32) wmma_gemm_kernel(
    const float* __restrict__ A, const float* __restrict__ B, float* __restrict__ C,
    int K, int lda, int ldb, int ldc, float scale, int ntiles)
{
    int tile = blockIdx.x;
    int mt = tile / ntiles, nt = tile % ntiles;
    int lane = threadIdx.x;
    int hi = lane >> 4, lm = lane & 15;
    const float* arow = A + (size_t)(mt*16 + lm)*lda;
    int col = nt*16 + lm;
    v8f acc = {};
    for (int kk = 0; kk < K; kk += 32) {
        v16h a = load_a_cols32(arow + kk, hi);
        v16h b = load_b_k32(B, ldb, kk, col, hi);
        acc = __builtin_amdgcn_wmma_f32_16x16x32_f16(false, a, false, b,
                                                     (short)0, acc, false, false);
    }
#pragma unroll
    for (int r = 0; r < 8; r++)
        C[(size_t)(mt*16 + r + (hi<<3))*ldc + col] = acc[r]*scale;
}

// ---------------------------------------------------------------------------
// Gate kernel with WMMA: g = sigmoid((s @ Wg)/sqrt(32)), s = in[:, :32].
// out[:, :32] = silu(s); l=1 block * g[:, :16]; l=2 block * g[:, 16:24].
// One wave handles 16 rows; requires n%16==0. Wg is (32,24).
// ---------------------------------------------------------------------------
__global__ void __launch_bounds__(32) gate_kernel(
    const float* __restrict__ in, const float* __restrict__ Wg,
    float* __restrict__ out)
{
    __shared__ float g[16*24];
    int lane = threadIdx.x;
    int hi = lane >> 4, lm = lane & 15;
    int rowbase = blockIdx.x * 16;

    v16h a = load_a_cols32(in + (size_t)(rowbase + lm)*HDIM, hi);
    v16h b0, b1;
    int c1 = 16 + lm;
    int c1c = (c1 < 24) ? c1 : 0;      // clamped address, select value below
#pragma unroll
    for (int h = 0; h < 16; h++) {
        int kb = h + (hi << 4);
        b0[h] = (_Float16)Wg[kb*24 + lm];
        float bv = Wg[kb*24 + c1c];
        b1[h] = (_Float16)((c1 < 24) ? bv : 0.0f);
    }
    v8f d0 = {}, d1 = {};
    d0 = __builtin_amdgcn_wmma_f32_16x16x32_f16(false, a, false, b0, (short)0, d0, false, false);
    d1 = __builtin_amdgcn_wmma_f32_16x16x32_f16(false, a, false, b1, (short)0, d1, false, false);
    const float inv = 0.17677669529663687f;      // 1/sqrt(32)
#pragma unroll
    for (int r = 0; r < 8; r++) {
        int rl = r + (hi << 3);
        g[rl*24 + lm] = 1.0f/(1.0f + __expf(-d0[r]*inv));
        if (lm < 8)
            g[rl*24 + 16 + lm] = 1.0f/(1.0f + __expf(-d1[r]*inv));
    }
    __syncthreads();
    for (int idx = lane; idx < 16*HDIM; idx += 32) {
        int rl = idx / HDIM, c = idx % HDIM;
        int row = rowbase + rl;
        float v = in[(size_t)row*HDIM + c];
        float o;
        if (c < 32)       o = v/(1.0f + __expf(-v));          // silu
        else if (c < 80)  o = v * g[rl*24 + (c-32)/3];        // l=1, u<16
        else              o = v * g[rl*24 + 16 + (c-80)/5];   // l=2, u<8
        out[(size_t)row*HDIM + c] = o;
    }
}

// ---------------------------------------------------------------------------
// Irrep-wise linear via WMMA: out[n,w,i] = sum_u in[n,u,i] W[u,w] / sqrt(m)
// l=0: (n,32)@(32,32)  -> 2 N-tiles, K=32 exact
// l=1: 3x (n,16)@(16,16) -> K padded 16->32 (pad indices are h>=8 in both lane halves,
//      i.e. compile-time static zeros, no guarded loads)
// l=2: 5x (n,8)@(8,8)    -> K padded 8->32, N padded 8->16
// One wave per 16 rows; requires n%16==0.
// ---------------------------------------------------------------------------
__global__ void __launch_bounds__(32) irlin_wmma_kernel(
    const float* __restrict__ in,
    const float* __restrict__ W0, const float* __restrict__ W1,
    const float* __restrict__ W2, float* __restrict__ out)
{
    int lane = threadIdx.x;
    int hi = lane >> 4, lm = lane & 15;
    int rowbase = blockIdx.x * 16;
    const float* arow = in + (size_t)(rowbase + lm)*HDIM;

    // ---- l = 0 : columns 0..31, two 16-wide output tiles, K = 32
    {
        v16h a = load_a_cols32(arow, hi);
#pragma unroll
        for (int nt = 0; nt < 2; nt++) {
            int col = nt*16 + lm;
            v16h b = load_b_k32(W0, 32, 0, col, hi);
            v8f d = {};
            d = __builtin_amdgcn_wmma_f32_16x16x32_f16(false, a, false, b,
                                                       (short)0, d, false, false);
#pragma unroll
            for (int r = 0; r < 8; r++)
                out[(size_t)(rowbase + r + (hi<<3))*HDIM + col]
                    = d[r]*0.17677669529663687f;   // 1/sqrt(32)
        }
    }

    // ---- l = 1 : for k in 0..2, A[m][u] = in[row, 32+u*3+k] (u<16), K pad 32
    for (int k = 0; k < 3; k++) {
        v16h a, b;
#pragma unroll
        for (int h = 0; h < 8; h++) {
            int u = h + (hi << 3);                       // 0..15
            a[h]   = (_Float16)arow[32 + u*3 + k];
            a[h+8] = (_Float16)0.0f;                     // k-index >= 16: pad
            float bv = W1[(h + (hi << 3))*16 + lm];      // kb = h + 8*hi
            b[h]   = (_Float16)bv;
            b[h+8] = (_Float16)0.0f;                     // kb >= 16: pad
        }
        v8f d = {};
        d = __builtin_amdgcn_wmma_f32_16x16x32_f16(false, a, false, b,
                                                   (short)0, d, false, false);
#pragma unroll
        for (int r = 0; r < 8; r++)
            out[(size_t)(rowbase + r + (hi<<3))*HDIM + 32 + lm*3 + k]
                = d[r]*0.25f;                            // 1/sqrt(16)
    }

    // ---- l = 2 : for k in 0..4, A[m][u] = in[row, 80+u*5+k] (u<8), K pad 32
    for (int k = 0; k < 5; k++) {
        v16h a, b;
        int lmc = lm & 7;
#pragma unroll
        for (int h = 0; h < 8; h++) {
            float av = arow[80 + h*5 + k];               // u = h (valid u<8)
            a[h]   = (_Float16)(hi ? 0.0f : av);
            a[h+8] = (_Float16)0.0f;
            float bv = W2[h*8 + lmc];
            b[h]   = (_Float16)((hi || lm >= 8) ? 0.0f : bv);
            b[h+8] = (_Float16)0.0f;
        }
        v8f d = {};
        d = __builtin_amdgcn_wmma_f32_16x16x32_f16(false, a, false, b,
                                                   (short)0, d, false, false);
        if (lm < 8) {
#pragma unroll
            for (int r = 0; r < 8; r++)
                out[(size_t)(rowbase + r + (hi<<3))*HDIM + 80 + lm*5 + k]
                    = d[r]*0.35355339059327373f;         // 1/sqrt(8)
        }
    }
}

// ---------------------------------------------------------------------------
// Fully-connected tensor product (fctp): LDS-staged two-stage contraction.
// ---------------------------------------------------------------------------
struct FPath {
    int t_off, w3j_off;
    int off1, off2;
    int m1, d1, m2, d2, m3, d3, i3;
    float norm;
};
struct WPtrs { const float* w[11]; };

// HIDDEN x HIDDEN -> HIDDEN paths (key-sorted, matches tp1/tpu leaf order)
__constant__ FPath PATHS_HH[11] = {
    {   0,   0,  0,  0, 32,1,32,1,32,1,0, 0.03125f      },  // 0.0.0
    {1024,   1,  0, 32, 32,1,16,3,16,3,1, 0.04419417382f},  // 0.1.1
    {2560,  10,  0, 80, 32,1, 8,5, 8,5,2, 0.0625f       },  // 0.2.2
    {3840,  35, 32,  0, 16,3,32,1,16,3,1, 0.04419417382f},  // 1.0.1
    {5376,  44, 32, 32, 16,3,16,3,32,1,0, 0.0625f       },  // 1.1.0
    {5632,  53, 32, 32, 16,3,16,3, 8,5,2, 0.0625f       },  // 1.1.2
    {6912,  98, 32, 80, 16,3, 8,5,16,3,1, 0.08838834765f},  // 1.2.1
    {7296, 143, 80,  0,  8,5,32,1, 8,5,2, 0.0625f       },  // 2.0.2
    {8576, 168, 80, 32,  8,5,16,3,16,3,1, 0.08838834765f},  // 2.1.1
    {8960, 213, 80, 80,  8,5, 8,5,32,1,0, 0.125f        },  // 2.2.0
    {9024, 238, 80, 80,  8,5, 8,5, 8,5,2, 0.125f        },  // 2.2.2
};
// HIDDEN x EDGE -> HIDDEN paths (key-sorted, matches tp2 leaf order)
__constant__ FPath PATHS_HE[7] = {
    {   0,   1,  0, 0, 32,1,1,3,16,3,1, 0.17677669530f},    // 0.0.1  (0,1,1)
    {  96,   0,  0, 3, 32,1,1,1,32,1,0, 0.17677669530f},    // 0.1.0  (0,0,0)
    { 128,  44, 32, 0, 16,3,1,3,32,1,0, 0.25f         },    // 1.0.0  (1,1,0)
    { 144,  53, 32, 0, 16,3,1,3, 8,5,2, 0.25f         },    // 1.0.2  (1,1,2)
    { 224,  35, 32, 3, 16,3,1,1,16,3,1, 0.25f         },    // 1.1.1  (1,0,1)
    { 272, 168, 80, 0,  8,5,1,3,16,3,1, 0.35355339059f},    // 2.0.1  (2,1,1)
    { 296, 143, 80, 3,  8,5,1,1, 8,5,2, 0.35355339059f},    // 2.1.2  (2,0,2)
};

template<int NP, int TSZ, int X2LEN>
__device__ void fctp_body(const FPath* P,
                          const float* __restrict__ x1, const int* __restrict__ idx1,
                          const float* __restrict__ x2, const int* __restrict__ idx2,
                          int x2stride, WPtrs Wp, const float* __restrict__ w3j,
                          float* __restrict__ out)
{
    __shared__ float sx1[HDIM];
    __shared__ float sx2[X2LEN];
    __shared__ float st[TSZ];
    int nidx = blockIdx.x;
    int r1 = idx1 ? idx1[nidx] : nidx;
    int r2 = idx2 ? idx2[nidx] : nidx;
    for (int i = threadIdx.x; i < HDIM;  i += blockDim.x) sx1[i] = x1[(size_t)r1*HDIM + i];
    for (int i = threadIdx.x; i < X2LEN; i += blockDim.x) sx2[i] = x2[(size_t)r2*x2stride + i];
    __syncthreads();
    // Stage 1: t[u,v,k] = sum_{i,j} x1[u,i] x2[v,j] C[i,j,k]
    for (int e = threadIdx.x; e < TSZ; e += blockDim.x) {
        int p = 0;
        while (p+1 < NP && e >= P[p+1].t_off) p++;
        FPath fp = P[p];
        int local = e - fp.t_off;
        int k  = local % fp.d3;
        int uv = local / fp.d3;
        int v  = uv % fp.m2;
        int u  = uv / fp.m2;
        const float* C = w3j + fp.w3j_off;
        float acc = 0.0f;
        for (int i = 0; i < fp.d1; i++) {
            float a = sx1[fp.off1 + u*fp.d1 + i];
            for (int j = 0; j < fp.d2; j++)
                acc += a * sx2[fp.off2 + v*fp.d2 + j] * C[(i*fp.d2 + j)*fp.d3 + k];
        }
        st[e] = acc;
    }
    __syncthreads();
    // Stage 2: out[w,k] = sum_p norm_p * sum_{u,v} t_p[u,v,k] W_p[u,v,w]
    for (int o = threadIdx.x; o < HDIM; o += blockDim.x) {
        int i3, w, k;
        if (o < 32)      { i3 = 0; w = o;        k = 0; }
        else if (o < 80) { i3 = 1; w = (o-32)/3; k = (o-32)%3; }
        else             { i3 = 2; w = (o-80)/5; k = (o-80)%5; }
        float acc = 0.0f;
        for (int p = 0; p < NP; p++) {
            if (P[p].i3 != i3) continue;
            FPath fp = P[p];
            const float* W = Wp.w[p];
            float s = 0.0f;
            int uvmax = fp.m1*fp.m2;
            for (int uv = 0; uv < uvmax; uv++)
                s += st[fp.t_off + uv*fp.d3 + k] * W[uv*fp.m3 + w];
            acc += fp.norm * s;
        }
        out[(size_t)nidx*HDIM + o] = acc;
    }
}

__global__ void __launch_bounds__(128) fctp_hh_kernel(
    const float* x1, const int* idx1, const float* x2, const int* idx2,
    WPtrs Wp, const float* w3j, float* out)
{
    fctp_body<11, 9344, HDIM>(PATHS_HH, x1, idx1, x2, idx2, HDIM, Wp, w3j, out);
}

__global__ void __launch_bounds__(128) fctp_he_kernel(
    const float* x1, const float* ea, WPtrs Wp, const float* w3j, float* out)
{
    fctp_body<7, 336, 4>(PATHS_HE, x1, nullptr, ea, nullptr, 4, Wp, w3j, out);
}

// ---------------------------------------------------------------------------
// Segment-sum / pooling kernels
// ---------------------------------------------------------------------------
__global__ void scatter_add_kernel(const float* __restrict__ m,
                                   const int* __restrict__ colIdx,
                                   float* __restrict__ agg)
{
    int idx = blockIdx.x*blockDim.x + threadIdx.x;
    if (idx >= NEDGES*HDIM) return;
    int e = idx / HDIM, o = idx % HDIM;
    atomicAdd(&agg[(size_t)colIdx[e]*HDIM + o], m[idx]);
}

__global__ void pool_kernel(const float* __restrict__ h,
                            const int* __restrict__ batch,
                            float* __restrict__ sums, float* __restrict__ cnt)
{
    int idx = blockIdx.x*blockDim.x + threadIdx.x;
    if (idx >= NNODES*32) return;
    int v = idx / 32, j = idx % 32;
    int g = batch[v];
    atomicAdd(&sums[g*32 + j], h[(size_t)v*HDIM + j]);
    if (j == 0) atomicAdd(&cnt[g], 1.0f);
}

__global__ void pool_div_kernel(const float* __restrict__ sums,
                                const float* __restrict__ cnt,
                                float* __restrict__ pooled)
{
    int idx = blockIdx.x*blockDim.x + threadIdx.x;
    if (idx >= NGRAPH*32) return;
    int g = idx / 32;
    pooled[idx] = sums[idx] / fmaxf(cnt[g], 1.0f);
}

// ---------------------------------------------------------------------------
// Host launcher
// ---------------------------------------------------------------------------
extern "C" void kernel_launch(void* const* d_in, const int* in_sizes, int n_in,
                              void* d_out, int out_size, void* d_ws, size_t ws_size,
                              hipStream_t stream)
{
    (void)in_sizes; (void)n_in; (void)out_size;
    const float* x      = (const float*)d_in[0];
    const float* pos    = (const float*)d_in[1];
    const int*   ei     = (const int*)d_in[2];
    const int*   batch  = (const int*)d_in[3];
    // params flattened as a JAX pytree (dict keys sorted):
    const float* finalW = (const float*)d_in[4];   // (32,512)
    const float* initW  = (const float*)d_in[5];   // (192,32)
    auto P = [&](int i)->const float* { return (const float*)d_in[i]; };

    // Workspace layout (floats)
    float* W3   = (float*)d_ws;                       // 512
    float* EA   = W3   + 512;                         // NEDGES*4
    float* H    = EA   + (size_t)NEDGES*4;            // NNODES*HDIM
    float* AGG  = H    + (size_t)NNODES*HDIM;
    float* UB   = AGG  + (size_t)NNODES*HDIM;
    float* TN   = UB   + (size_t)NNODES*HDIM;
    float* MA   = TN   + (size_t)NNODES*HDIM;         // NEDGES*HDIM
    float* MB   = MA   + (size_t)NEDGES*HDIM;
    float* SUMS = MB   + (size_t)NEDGES*HDIM;         // 64*32
    float* CNT  = SUMS + NGRAPH*32;                   // 64
    float* POOL = CNT  + NGRAPH;                      // 64*32
    size_t need = (size_t)(POOL + NGRAPH*32 - W3) * sizeof(float);
    if (ws_size < need) return;

    // Constants & edge attributes
    w3j_init_kernel<<<1, 32, 0, stream>>>(W3);
    edge_attr_kernel<<<(NEDGES+255)/256, 256, 0, stream>>>(pos, ei, EA);

    // h = [x @ init_W / sqrt(192), zeros]
    zero_kernel<<<(NNODES*HDIM+255)/256, 256, 0, stream>>>(H, NNODES*HDIM);
    wmma_gemm_kernel<<<(NNODES/16)*2, 32, 0, stream>>>(
        x, initW, H, 192, 192, 32, HDIM, 0.07216878364f, 2);

    for (int L = 0; L < 3; L++) {
        int base = 6 + 37*L;
        const float* linm0 = P(base+0), *linm1 = P(base+1), *linm2 = P(base+2);
        const float* linu0 = P(base+3), *linu1 = P(base+4), *linu2 = P(base+5);
        WPtrs tp1{}; for (int i=0;i<11;i++) tp1.w[i] = P(base+6+i);
        WPtrs tp2{}; for (int i=0;i<7; i++) tp2.w[i] = P(base+17+i);
        WPtrs tpu{}; for (int i=0;i<11;i++) tpu.w[i] = P(base+24+i);
        const float* wgm = P(base+35);
        const float* wgu = P(base+36);

        // m = fctp(h[row], h[col], tp1)
        fctp_hh_kernel<<<NEDGES, 128, 0, stream>>>(H, ei, H, ei+NEDGES, tp1, W3, MA);
        // m = fctp(m, ea, tp2)
        fctp_he_kernel<<<NEDGES, 128, 0, stream>>>(MA, EA, tp2, W3, MB);
        // m = irlin(gate(m, wg_m), lin_m)
        gate_kernel<<<NEDGES/16, 32, 0, stream>>>(MB, wgm, MA);
        irlin_wmma_kernel<<<NEDGES/16, 32, 0, stream>>>(MA, linm0, linm1, linm2, MB);
        // agg = segment_sum(m, col)
        zero_kernel<<<(NNODES*HDIM+255)/256, 256, 0, stream>>>(AGG, NNODES*HDIM);
        scatter_add_kernel<<<(NEDGES*HDIM+255)/256, 256, 0, stream>>>(MB, ei+NEDGES, AGG);
        // u = fctp(h, agg, tpu);  h = irlin(gate(u, wg_u), lin_u)
        fctp_hh_kernel<<<NNODES, 128, 0, stream>>>(H, nullptr, AGG, nullptr, tpu, W3, UB);
        gate_kernel<<<NNODES/16, 32, 0, stream>>>(UB, wgu, TN);
        irlin_wmma_kernel<<<NNODES/16, 32, 0, stream>>>(TN, linu0, linu1, linu2, H);
    }

    // mean-pool scalars per graph, then final projection
    zero_kernel<<<(NGRAPH*32+255)/256, 256, 0, stream>>>(SUMS, NGRAPH*32);
    zero_kernel<<<1, 64, 0, stream>>>(CNT, NGRAPH);
    pool_kernel<<<(NNODES*32+255)/256, 256, 0, stream>>>(H, batch, SUMS, CNT);
    pool_div_kernel<<<(NGRAPH*32+255)/256, 256, 0, stream>>>(SUMS, CNT, POOL);
    wmma_gemm_kernel<<<(NGRAPH/16)*(512/16), 32, 0, stream>>>(
        POOL, finalW, (float*)d_out, 32, 32, 512, 512,
        0.17677669529663687f, 512/16);
}
